// SharedBackbone_84361747628703
// MI455X (gfx1250) — matrix operand order repaired and checked by reading, not compiled, not run on hardware
//
#include <hip/hip_runtime.h>
#include <hip/hip_bf16.h>

#define HID    256
#define HEADS  8
#define CPH    32
#define INCH   128

typedef __attribute__((ext_vector_type(16))) __bf16 v16bf;
typedef __attribute__((ext_vector_type(8)))  float  v8f;

// ---------------------------------------------------------------------------
// Weight packing: fp32 [K, Nout] row-major -> bf16 B-fragment panels.
// Panel (i = k-tile of 32, j = n-tile of 16), stored per-lane:
//   lane l holds 16 contiguous bf16 = B[k = i*32 + (l>>4)*16 + e][n = j*16 + (l&15)]
// a contiguous k order, the same one the A fragments use in the GEMM kernel.
// ---------------------------------------------------------------------------
__global__ void pack_b_kernel(const float* __restrict__ W, __bf16* __restrict__ Bp,
                              int K, int Nout) {
    int pk = K >> 5, pn = Nout >> 4;
    int total = pk * pn * 32;
    int t = blockIdx.x * blockDim.x + threadIdx.x;
    if (t >= total) return;
    int lane  = t & 31;
    int panel = t >> 5;
    int i = panel % pk;            // k tile
    int j = panel / pk;            // n tile
    int kbase = i * 32 + (lane >> 4) * 16;
    int n     = j * 16 + (lane & 15);
    __bf16* dst = Bp + (size_t)t * 16;
#pragma unroll
    for (int e = 0; e < 16; ++e)
        dst[e] = (__bf16)W[(size_t)(kbase + e) * Nout + n];
}

// ---------------------------------------------------------------------------
// C[M,NOUT] = A[M,K] @ B + bias via v_wmma_f32_16x16x32_bf16.
// Block = 4 waves; wave w owns row tile m0 = (blockIdx*4+w)*16. All K/32 A
// fragments are loaded+converted ONCE into registers, then B panels for all
// NOUT/16 column tiles stream against them (B tiny, L2-hot). Bias is never
// null (caller passes a zero vector) and is preloaded; stores use a uniform
// full-tile fast path so the hot loop has no exec-mask churn.
// ---------------------------------------------------------------------------
template<int K, int NOUT>
__global__ void gemm_wmma_kernel(const float* __restrict__ A, const __bf16* __restrict__ Bp,
                                 const float* __restrict__ bias, float* __restrict__ C,
                                 int M) {
    constexpr int PK = K / 32;
    constexpr int PN = NOUT / 16;
    const int lane = threadIdx.x & 31;
    const int wave = threadIdx.x >> 5;
    const int m0   = (blockIdx.x * 4 + wave) << 4;
    if (m0 >= M) return;
    const int half = lane >> 4;
    const int nc   = lane & 15;
    int row = m0 + nc;
    if (row >= M) row = M - 1;                    // clamp partial tile
    const float* __restrict__ arow = A + (size_t)row * K;

    // --- load + convert all A fragments once ---
    v16bf afrag[PK];
#pragma unroll
    for (int i = 0; i < PK; ++i) {
        const float4* p0 = (const float4*)(arow + i * 32 + half * 16);
        const float4* p1 = (const float4*)(arow + i * 32 + half * 16 + 8);
        float4 f0 = p0[0], f1 = p0[1], f2 = p1[0], f3 = p1[1];
        v16bf a;
        a[0]  = (__bf16)f0.x; a[1]  = (__bf16)f0.y; a[2]  = (__bf16)f0.z; a[3]  = (__bf16)f0.w;
        a[4]  = (__bf16)f1.x; a[5]  = (__bf16)f1.y; a[6]  = (__bf16)f1.z; a[7]  = (__bf16)f1.w;
        a[8]  = (__bf16)f2.x; a[9]  = (__bf16)f2.y; a[10] = (__bf16)f2.z; a[11] = (__bf16)f2.w;
        a[12] = (__bf16)f3.x; a[13] = (__bf16)f3.y; a[14] = (__bf16)f3.z; a[15] = (__bf16)f3.w;
        afrag[i] = a;
    }

    // --- preload bias (never null) ---
    float bvs[PN];
#pragma unroll
    for (int j = 0; j < PN; ++j) bvs[j] = bias[j * 16 + nc];

    const int  crow0 = m0 + half * 8;
    const bool full  = (m0 + 16 <= M);            // uniform across wave
    for (int j = 0; j < PN; ++j) {
        v8f acc = {0.f, 0.f, 0.f, 0.f, 0.f, 0.f, 0.f, 0.f};
        const v16bf* __restrict__ bptr =
            (const v16bf*)(Bp + ((size_t)j * PK * 32 + lane) * 16);
#pragma unroll
        for (int i = 0; i < PK; ++i)
            acc = __builtin_amdgcn_wmma_f32_16x16x32_bf16(
                      false, afrag[i], false, bptr[i * 32], (short)0, acc, false, false);
        float  bv   = bvs[j];
        float* cptr = C + (size_t)crow0 * NOUT + j * 16 + nc;
        if (full) {
#pragma unroll
            for (int r = 0; r < 8; ++r) { cptr[0] = acc[r] + bv; cptr += NOUT; }
        } else {
#pragma unroll
            for (int r = 0; r < 8; ++r) {
                if (crow0 + r < M) cptr[0] = acc[r] + bv;
                cptr += NOUT;
            }
        }
    }
}

// ---------------------------------------------------------------------------
// Attention scores: a_src[n,h] = <h[n,h,:], att_src[h,:]>, same for dst.
// ---------------------------------------------------------------------------
__global__ void att_scores_kernel(const float* __restrict__ h,
                                  const float* __restrict__ att_src,
                                  const float* __restrict__ att_dst,
                                  float* __restrict__ asrc, float* __restrict__ adst, int N) {
    int t = blockIdx.x * blockDim.x + threadIdx.x;
    if (t >= N * HEADS) return;
    int n = t >> 3, hd = t & 7;
    const float* hp = h + (size_t)n * HID + hd * CPH;
    const float* as = att_src + hd * CPH;
    const float* ad = att_dst + hd * CPH;
    float s = 0.f, d = 0.f;
#pragma unroll
    for (int c = 0; c < CPH; ++c) { float v = hp[c]; s += v * as[c]; d += v * ad[c]; }
    asrc[t] = s; adst[t] = d;
}

__global__ void fill_kernel(float* __restrict__ p, float v, size_t n) {
    size_t i = (size_t)blockIdx.x * blockDim.x + threadIdx.x;
    if (i < n) p[i] = v;
}

__device__ __forceinline__ void atomicMaxFloat(float* addr, float v) {
    if (v >= 0.f) atomicMax((int*)addr, __float_as_int(v));
    else          atomicMin((unsigned int*)addr, __float_as_uint(v));
}

// pass 1: leaky-relu alpha, segment max over dst
__global__ void edge_max_kernel(const int* __restrict__ ei,
                                const float* __restrict__ asrc, const float* __restrict__ adst,
                                float* __restrict__ amax, int E, int N) {
    int t = blockIdx.x * blockDim.x + threadIdx.x;
    int total = (E + N) * HEADS;
    if (t >= total) return;
    int e = t >> 3, hd = t & 7;
    int s, d;
    if (e < E) { s = ei[e]; d = ei[E + e]; } else { s = e - E; d = s; }
    float a = asrc[s * HEADS + hd] + adst[d * HEADS + hd];
    a = fmaxf(a, 0.2f * a);                      // leaky_relu(0.2)
    atomicMaxFloat(&amax[d * HEADS + hd], a);
}

// pass 2: ealpha = exp(alpha - max), accumulate denom
__global__ void edge_exp_kernel(const int* __restrict__ ei,
                                const float* __restrict__ asrc, const float* __restrict__ adst,
                                const float* __restrict__ amax,
                                float* __restrict__ denom, float* __restrict__ ealpha,
                                int E, int N) {
    int t = blockIdx.x * blockDim.x + threadIdx.x;
    int total = (E + N) * HEADS;
    if (t >= total) return;
    int e = t >> 3, hd = t & 7;
    int s, d;
    if (e < E) { s = ei[e]; d = ei[E + e]; } else { s = e - E; d = s; }
    float a = asrc[s * HEADS + hd] + adst[d * HEADS + hd];
    a = fmaxf(a, 0.2f * a);
    float ea = __expf(a - amax[d * HEADS + hd]);
    ealpha[t] = ea;
    atomicAdd(&denom[d * HEADS + hd], ea);
}

// pass 2b: ealpha -> attn = ealpha / (denom[dst] + eps)   (once per edge-head)
__global__ void edge_norm_kernel(const int* __restrict__ ei, const float* __restrict__ denom,
                                 float* __restrict__ ealpha, int E, int N) {
    int t = blockIdx.x * blockDim.x + threadIdx.x;
    int total = (E + N) * HEADS;
    if (t >= total) return;
    int e = t >> 3, hd = t & 7;
    int d = (e < E) ? ei[E + e] : (e - E);
    ealpha[t] = ealpha[t] / (denom[d * HEADS + hd] + 1e-16f);
}

// pass 3: out[dst] += h[src] * attn  (thread = edge, head, 8-channel slice)
__global__ void edge_agg_kernel(const int* __restrict__ ei, const float* __restrict__ h,
                                const float* __restrict__ attn, float* __restrict__ out,
                                int E, int N) {
    int t = blockIdx.x * blockDim.x + threadIdx.x;
    int total = (E + N) * 32;                    // 8 heads * 4 slices
    if (t >= total) return;
    int e = t >> 5, sub = t & 31;
    int hd = sub >> 2, q = sub & 3;
    int s, d;
    if (e < E) { s = ei[e]; d = ei[E + e]; } else { s = e - E; d = s; }
    float at = attn[e * HEADS + hd];
    const float* hp = h + (size_t)s * HID + hd * CPH + q * 8;
    float* op       = out + (size_t)d * HID + hd * CPH + q * 8;
#pragma unroll
    for (int c = 0; c < 8; ++c) atomicAdd(&op[c], hp[c] * at);
}

// column sums & sumsq partials (blockDim.x == Ncols)
__global__ void col_stats_kernel(const float* __restrict__ x, float* __restrict__ stats,
                                 int Nrows, int Ncols, int rowsPerBlock) {
    int ch = threadIdx.x;
    int r0 = blockIdx.x * rowsPerBlock;
    int rend = r0 + rowsPerBlock; if (rend > Nrows) rend = Nrows;
    float s = 0.f, sq = 0.f;
    for (int r = r0; r < rend; ++r) {
        float v = x[(size_t)r * Ncols + ch];
        s += v; sq += v * v;
    }
    atomicAdd(&stats[ch], s);
    atomicAdd(&stats[Ncols + ch], sq);
}

// BN (biased var) -> scale/shift in stats[2N..4N)
__global__ void bn_finalize_kernel(float* __restrict__ stats, const float* __restrict__ g,
                                   const float* __restrict__ b, int Nrows, int Ncols) {
    int ch = blockIdx.x * blockDim.x + threadIdx.x;
    if (ch >= Ncols) return;
    float inv = 1.f / (float)Nrows;
    float mean = stats[ch] * inv;
    float var  = stats[Ncols + ch] * inv - mean * mean;
    float sc = g[ch] * rsqrtf(var + 1e-5f);
    stats[2 * Ncols + ch] = sc;
    stats[3 * Ncols + ch] = b[ch] - mean * sc;
}

// y = elu(scale*x + shift) in place
__global__ void bn_elu_kernel(float* __restrict__ x, const float* __restrict__ stats,
                              size_t n, int Ncols) {
    size_t i = (size_t)blockIdx.x * blockDim.x + threadIdx.x;
    if (i >= n) return;
    int ch = (int)(i % Ncols);
    float v = x[i] * stats[2 * Ncols + ch] + stats[3 * Ncols + ch];
    x[i] = v > 0.f ? v : (__expf(v) - 1.f);
}

__global__ void add3_kernel(float* __restrict__ dst, const float* __restrict__ a,
                            const float* __restrict__ b, const float* __restrict__ c, size_t n) {
    size_t i = (size_t)blockIdx.x * blockDim.x + threadIdx.x;
    if (i >= n) return;
    float v = a[i] + b[i];
    if (c) v += c[i];
    dst[i] = v;
}

// final column standardize: std with ddof=1 -> scale/shift
__global__ void std_finalize_kernel(float* __restrict__ stats, int Nrows, int Ncols) {
    int ch = blockIdx.x * blockDim.x + threadIdx.x;
    if (ch >= Ncols) return;
    float mean = stats[ch] / (float)Nrows;
    float var  = (stats[Ncols + ch] - (float)Nrows * mean * mean) / (float)(Nrows - 1);
    float sd = sqrtf(fmaxf(var, 0.f));
    float sc = 1.f / (sd + 1e-6f);
    stats[2 * Ncols + ch] = sc;
    stats[3 * Ncols + ch] = -mean * sc;
}

// one block per row: standardize then L2-normalize the row
__global__ void final_rows_kernel(const float* __restrict__ x, const float* __restrict__ stats,
                                  float* __restrict__ out) {
    __shared__ float red[HID];
    int r = blockIdx.x, ch = threadIdx.x;
    float v = x[(size_t)r * HID + ch] * stats[2 * HID + ch] + stats[3 * HID + ch];
    red[ch] = v * v;
    __syncthreads();
    for (int s = HID / 2; s > 0; s >>= 1) {
        if (ch < s) red[ch] += red[ch + s];
        __syncthreads();
    }
    float nrm = fmaxf(sqrtf(red[0]), 1e-12f);
    out[(size_t)r * HID + ch] = v / nrm;
}

// ---------------------------------------------------------------------------
extern "C" void kernel_launch(void* const* d_in, const int* in_sizes, int n_in,
                              void* d_out, int out_size, void* d_ws, size_t ws_size,
                              hipStream_t stream) {
    (void)n_in; (void)out_size; (void)ws_size;
    const float* x_in = (const float*)d_in[0];
    const int*   ei   = (const int*)d_in[1];
    const int N    = in_sizes[0] / INCH;
    const int E    = in_sizes[1] / 2;
    const int Etot = E + N;

    // per-layer params: 0 W,1 att_src,2 att_dst,3 res_W,4 bias,5 bn_g,6 bn_b,
    // 7 dw,8 db,9 dg,10 dbeta,11 uw,12 ub,13 ug,14 ubeta,[15 sw,16 sb layer0 only]
    const float* P[3][17];
    int idx = 2;
    for (int li = 0; li < 3; ++li) {
        int cnt = (li == 0) ? 17 : 15;
        for (int k = 0; k < cnt; ++k) P[li][k] = (const float*)d_in[idx++];
        if (li != 0) { P[li][15] = nullptr; P[li][16] = nullptr; }
    }

    // workspace layout (floats)
    float* ws = (float*)d_ws;
    size_t o = 0;
    float* xbuf  = ws + o; o += (size_t)N * HID;
    float* hbuf  = ws + o; o += (size_t)N * HID;
    float* resb  = ws + o; o += (size_t)N * HID;
    float* ubuf  = ws + o; o += (size_t)N * HID;
    float* dbuf  = ws + o; o += (size_t)N * (HID / 4);
    float* asrc  = ws + o; o += (size_t)N * HEADS;
    float* adst  = ws + o; o += (size_t)N * HEADS;
    float* amax  = ws + o; o += (size_t)N * HEADS;
    float* deno  = ws + o; o += (size_t)N * HEADS;
    float* ealp  = ws + o; o += (size_t)Etot * HEADS;
    float* stats = ws + o; o += 1024;
    float* zbias = ws + o; o += HID;              // zero "bias" for bias-free GEMMs
    __bf16* packb = (__bf16*)(ws + o);            // up to 256*256 bf16

    const int TPB = 256;
    auto blocks = [](size_t n) { return (unsigned)((n + 255) / 256); };

    fill_kernel<<<blocks(HID), TPB, 0, stream>>>(zbias, 0.f, (size_t)HID);

    auto packW = [&](const float* Wp, int K, int Nout) {
        int total = (K >> 5) * (Nout >> 4) * 32;
        pack_b_kernel<<<blocks(total), TPB, 0, stream>>>(Wp, packb, K, Nout);
    };
    const unsigned gemm_grid = (unsigned)((N + 63) / 64);
    auto gemm = [&](const float* Ap, const float* bias, float* Cp, int K, int Nout) {
        if (!bias) bias = zbias;
        if (K == INCH && Nout == HID)
            gemm_wmma_kernel<INCH, HID><<<gemm_grid, 128, 0, stream>>>(Ap, packb, bias, Cp, N);
        else if (K == HID && Nout == HID)
            gemm_wmma_kernel<HID, HID><<<gemm_grid, 128, 0, stream>>>(Ap, packb, bias, Cp, N);
        else if (K == HID && Nout == HID / 4)
            gemm_wmma_kernel<HID, HID / 4><<<gemm_grid, 128, 0, stream>>>(Ap, packb, bias, Cp, N);
        else
            gemm_wmma_kernel<HID / 4, HID><<<gemm_grid, 128, 0, stream>>>(Ap, packb, bias, Cp, N);
    };
    auto bn_block = [&](float* buf, const float* g, const float* b, int Ncols) {
        fill_kernel<<<blocks(2 * Ncols), TPB, 0, stream>>>(stats, 0.f, (size_t)2 * Ncols);
        col_stats_kernel<<<(N + 127) / 128, Ncols, 0, stream>>>(buf, stats, N, Ncols, 128);
        bn_finalize_kernel<<<1, Ncols, 0, stream>>>(stats, g, b, N, Ncols);
        bn_elu_kernel<<<blocks((size_t)N * Ncols), TPB, 0, stream>>>(buf, stats, (size_t)N * Ncols, Ncols);
    };

    for (int li = 0; li < 3; ++li) {
        const int K_in = (li == 0) ? INCH : HID;
        const float* xin = (li == 0) ? x_in : xbuf;
        const float** p = P[li];

        // h = xin @ W
        packW(p[0], K_in, HID);
        gemm(xin, nullptr, hbuf, K_in, HID);
        // res = xin @ res_W + bias   (aggregation accumulates into this)
        packW(p[3], K_in, HID);
        gemm(xin, p[4], resb, K_in, HID);

        // attention softmax over in-edges of dst
        att_scores_kernel<<<blocks((size_t)N * HEADS), TPB, 0, stream>>>(hbuf, p[1], p[2], asrc, adst, N);
        fill_kernel<<<blocks((size_t)N * HEADS), TPB, 0, stream>>>(amax, -3.402823466e38f, (size_t)N * HEADS);
        fill_kernel<<<blocks((size_t)N * HEADS), TPB, 0, stream>>>(deno, 0.f, (size_t)N * HEADS);
        edge_max_kernel<<<blocks((size_t)Etot * HEADS), TPB, 0, stream>>>(ei, asrc, adst, amax, E, N);
        edge_exp_kernel<<<blocks((size_t)Etot * HEADS), TPB, 0, stream>>>(ei, asrc, adst, amax, deno, ealp, E, N);
        edge_norm_kernel<<<blocks((size_t)Etot * HEADS), TPB, 0, stream>>>(ei, deno, ealp, E, N);
        edge_agg_kernel<<<blocks((size_t)Etot * 32), TPB, 0, stream>>>(ei, hbuf, ealp, resb, E, N);

        // x1 = elu(bn(res))   (in place in resb)
        bn_block(resb, p[5], p[6], HID);
        // d = elu(bn(x1 @ dw + db))
        packW(p[7], HID, HID / 4);
        gemm(resb, p[8], dbuf, HID, HID / 4);
        bn_block(dbuf, p[9], p[10], HID / 4);
        // u = elu(bn(d @ uw + ub))
        packW(p[11], HID / 4, HID);
        gemm(dbuf, p[12], ubuf, HID / 4, HID);
        bn_block(ubuf, p[13], p[14], HID);

        // x = x1 + u (+ skip)
        if (li == 0) {
            packW(p[15], INCH, HID);              // skip proj into hbuf (h no longer needed)
            gemm(x_in, p[16], hbuf, INCH, HID);
            add3_kernel<<<blocks((size_t)N * HID), TPB, 0, stream>>>(xbuf, resb, ubuf, hbuf, (size_t)N * HID);
        } else {
            add3_kernel<<<blocks((size_t)N * HID), TPB, 0, stream>>>(xbuf, resb, ubuf, xbuf, (size_t)N * HID);
        }
    }

    // final: column standardize (ddof=1) then per-row L2 normalize
    fill_kernel<<<blocks(512), TPB, 0, stream>>>(stats, 0.f, 512);
    col_stats_kernel<<<(N + 127) / 128, HID, 0, stream>>>(xbuf, stats, N, HID, 128);
    std_finalize_kernel<<<1, HID, 0, stream>>>(stats, N, HID);
    final_rows_kernel<<<N, HID, 0, stream>>>(xbuf, stats, (float*)d_out);
}
